// CapsuleLayer_16088947491261
// MI455X (gfx1250) — matrix operand order repaired and checked by reading, not compiled, not run on hardware
//
#include <hip/hip_runtime.h>
#include <math.h>

typedef __attribute__((ext_vector_type(16))) _Float16 v16h;
typedef __attribute__((ext_vector_type(8)))  float    v8f;

#define B_   64
#define O_   32
#define I_   2048
#define DIN  16
#define DOUT 32
#define NCH  32
#define ICH  (I_ / NCH)   // 64 i's per routing block

union V16H { v16h v; _Float16 h[16]; };
union V8F  { v8f  v; float    f[8];  };

// ---------------------------------------------------------------------------
// Kernel 1: u_hat[b,i,o,d] = sum_c W[o,i,d,c] * x[b,i,c]   (stored as fp16)
// One block per i (2048 blocks, 8 waves). Wave w handles o = w + 8*{0..3}.
// WMMA: A = W-tile (M=d(16), K=c(16, zero-padded to 32)),
//       B = x-tile (K=c, N=b(16)), D = f32 [d x b]. 2 m-tiles x 4 n-tiles.
// ---------------------------------------------------------------------------
__global__ __launch_bounds__(256)
void uhat_gemm(const float* __restrict__ x, const float* __restrict__ W,
               _Float16* __restrict__ uhat)
{
    const int i    = blockIdx.x;
    const int tid  = threadIdx.x;
    const int wave = tid >> 5;
    const int lane = tid & 31;

    __shared__ _Float16 xs[B_][DIN];          // 2 KB: x[:, i, :] as fp16

    for (int p = tid; p < B_ * DIN; p += 256) {
        int b = p >> 4, c = p & 15;
        xs[b][c] = (_Float16)x[((size_t)b * I_ + i) * DIN + c];
    }
    __syncthreads();

    // B fragments (shared by every o this wave handles).
    // 16-bit B 32x16: lanes 0-15 hold K=0..7, lanes 16-31 K=8..15 in elems 0-7 (valid c); elems 8-15 K=16..31 (pad=0).
    V16H bfrag[4];
    for (int nt = 0; nt < 4; ++nt) {
        V16H f;
        #pragma unroll
        for (int e = 0; e < 16; ++e) f.h[e] = (_Float16)0.f;
        {
            int b = 16 * nt + (lane & 15);   // N index = b
            #pragma unroll
            for (int e = 0; e < 8; ++e) f.h[e] = xs[b][((lane < 16) ? 0 : 8) + e];
        }
        bfrag[nt] = f;
    }

    const int row_in = lane & 15;
    const int cbase  = (lane < 16) ? 0 : 8;  // A layout: lanes>=16 hold K=8..15

    for (int oo = 0; oo < 4; ++oo) {
        const int o = wave + 8 * oo;

        // A fragments from W[o,i,d,c] rows (contiguous 8 floats per lane).
        V16H afrag[2];
        #pragma unroll
        for (int mt = 0; mt < 2; ++mt) {
            const float* wp =
                &W[(((size_t)o * I_ + i) * DOUT + (16 * mt + row_in)) * DIN + cbase];
            V16H f;
            #pragma unroll
            for (int e = 0; e < 8; ++e) f.h[e] = (_Float16)wp[e];
            #pragma unroll
            for (int e = 8; e < 16; ++e) f.h[e] = (_Float16)0.f;  // K=16..31 pad
            afrag[mt] = f;
        }
        if (oo < 3)  // stream-prefetch next o's W slice
            __builtin_prefetch(&W[(((size_t)(o + 8) * I_ + i) * DOUT) * DIN], 0, 1);

        V8F acc[2][4];
        #pragma unroll
        for (int mt = 0; mt < 2; ++mt)
            #pragma unroll
            for (int nt = 0; nt < 4; ++nt) {
                #pragma unroll
                for (int r = 0; r < 8; ++r) acc[mt][nt].f[r] = 0.f;
                acc[mt][nt].v = __builtin_amdgcn_wmma_f32_16x16x32_f16(
                    false, afrag[mt].v, false, bfrag[nt].v,
                    (short)0, acc[mt][nt].v, false, false);
            }

        // D layout: VGPR r, lanes 0-15 -> M=r (d), lanes 16-31 -> M=8+r; N=lane%16 (b).
        // Each lane owns 8 consecutive d's of one b -> one 16B fp16 store.
        #pragma unroll
        for (int mt = 0; mt < 2; ++mt)
            #pragma unroll
            for (int nt = 0; nt < 4; ++nt) {
                int b  = 16 * nt + (lane & 15);
                int d0 = 16 * mt + ((lane < 16) ? 0 : 8);
                union { uint4 q; _Float16 h[8]; } pk;
                #pragma unroll
                for (int r = 0; r < 8; ++r) pk.h[r] = (_Float16)acc[mt][nt].f[r];
                *(uint4*)&uhat[(((size_t)b * I_ + i) * O_ + o) * DOUT + d0] = pk.q;
            }
    }
}

// ---------------------------------------------------------------------------
// Kernel 2: one fused routing pass. Block = (chunk, b); 8 warps x 8 i's each.
// Lane index == capsule o (wave32). Per i: agreement with v_prev, b_ij update,
// softmax over o across the 32 lanes, accumulate partial s in registers.
// Deterministic: fixed shuffle trees + fixed-order LDS cross-warp reduction.
// ---------------------------------------------------------------------------
__global__ __launch_bounds__(256)
void routing_pass(const _Float16* __restrict__ uhat, float* __restrict__ b_ws,
                  const float* __restrict__ v_prev, float* __restrict__ s_part,
                  int iter)
{
    const int chunk = blockIdx.x;
    const int b     = blockIdx.y;
    const int tid   = threadIdx.x;
    const int warp  = tid >> 5;
    const int o     = tid & 31;

    __shared__ float v_l[O_][DOUT];        // 4 KB  : v from previous iteration
    __shared__ float sred[8][O_][DOUT];    // 32 KB : cross-warp s reduction

    if (iter > 0) {
        for (int p = tid; p < O_ * DOUT; p += 256)
            (&v_l[0][0])[p] = v_prev[(size_t)b * O_ * DOUT + p];
    }
    __syncthreads();

    float sacc[DOUT];
    #pragma unroll
    for (int d = 0; d < DOUT; ++d) sacc[d] = 0.f;

    for (int k = 0; k < 8; ++k) {
        const int i = chunk * ICH + warp * 8 + k;
        const uint4* up =
            (const uint4*)&uhat[(((size_t)b * I_ + i) * O_ + o) * DOUT];
        union { uint4 q[4]; _Float16 h[DOUT]; } u;
        #pragma unroll
        for (int j = 0; j < 4; ++j) u.q[j] = up[j];
        if (k < 7)
            __builtin_prefetch((const char*)up + (size_t)O_ * DOUT * 2, 0, 1);

        float c;
        if (iter == 0) {
            c = 1.0f / (float)O_;          // softmax of zeros
        } else {
            float agr = 0.f;
            #pragma unroll
            for (int d = 0; d < DOUT; ++d) agr += (float)u.h[d] * v_l[o][d];
            const size_t bidx = ((size_t)b * I_ + i) * O_ + o;
            float bv;
            if (iter == 1) { bv = agr; b_ws[bidx] = bv; }     // b was 0
            else           { bv = b_ws[bidx] + agr; }          // last iter: no write
            float m = bv;
            #pragma unroll
            for (int s = 16; s > 0; s >>= 1) m = fmaxf(m, __shfl_xor(m, s, 32));
            float e = __expf(bv - m);
            float ss = e;
            #pragma unroll
            for (int s = 16; s > 0; s >>= 1) ss += __shfl_xor(ss, s, 32);
            c = e / ss;
        }
        #pragma unroll
        for (int d = 0; d < DOUT; ++d) sacc[d] += c * (float)u.h[d];
    }

    #pragma unroll
    for (int d = 0; d < DOUT; ++d) sred[warp][o][d] = sacc[d];
    __syncthreads();

    for (int p = tid; p < O_ * DOUT; p += 256) {
        int oo = p >> 5, dd = p & 31;
        float t = 0.f;
        #pragma unroll
        for (int w = 0; w < 8; ++w) t += sred[w][oo][dd];   // fixed order
        s_part[(((size_t)b * NCH + chunk) * O_ + oo) * DOUT + dd] = t;
    }
}

// ---------------------------------------------------------------------------
// Kernel 3: sum s over chunks, squash, write v (or final output).
// One wave per (b,o); lane = d.
// ---------------------------------------------------------------------------
__global__ __launch_bounds__(32)
void reduce_squash(const float* __restrict__ s_part, float* __restrict__ v_out)
{
    const int bo = blockIdx.x;          // b*O_ + o
    const int b  = bo >> 5, o = bo & 31;
    const int d  = threadIdx.x;

    float s = 0.f;
    for (int ch = 0; ch < NCH; ++ch)    // fixed order -> deterministic
        s += s_part[(((size_t)b * NCH + ch) * O_ + o) * DOUT + d];

    float sq = s * s;
    #pragma unroll
    for (int sh = 16; sh > 0; sh >>= 1) sq += __shfl_xor(sq, sh, 32);
    const float nsq   = sq;
    const float scale = (nsq / (1.f + nsq)) / sqrtf(nsq + 1e-9f);
    v_out[(size_t)bo * DOUT + d] = scale * s;
}

// ---------------------------------------------------------------------------
extern "C" void kernel_launch(void* const* d_in, const int* in_sizes, int n_in,
                              void* d_out, int out_size, void* d_ws, size_t ws_size,
                              hipStream_t stream)
{
    (void)in_sizes; (void)n_in; (void)out_size; (void)ws_size;
    const float* x = (const float*)d_in[0];   // [B, I, DIN] f32
    const float* W = (const float*)d_in[1];   // [O, I, DOUT, DIN] f32
    float* out = (float*)d_out;               // [B, O, DOUT] f32

    char* ws = (char*)d_ws;
    const size_t UHAT_BYTES = (size_t)B_ * I_ * O_ * DOUT * sizeof(_Float16); // 256 MB
    const size_t BIJ_BYTES  = (size_t)B_ * I_ * O_ * sizeof(float);           // 16 MB
    const size_t SP_BYTES   = (size_t)B_ * NCH * O_ * DOUT * sizeof(float);   // 8 MB
    _Float16* uhat   = (_Float16*)ws;
    float*    b_ws   = (float*)(ws + UHAT_BYTES);
    float*    s_part = (float*)(ws + UHAT_BYTES + BIJ_BYTES);
    float*    v_ws   = (float*)(ws + UHAT_BYTES + BIJ_BYTES + SP_BYTES);

    uhat_gemm<<<I_, 256, 0, stream>>>(x, W, uhat);

    for (int it = 0; it < 3; ++it) {
        routing_pass<<<dim3(NCH, B_), 256, 0, stream>>>(uhat, b_ws, v_ws, s_part, it);
        reduce_squash<<<B_ * O_, 32, 0, stream>>>(s_part, (it == 2) ? out : v_ws);
    }
}
